// WeedLayer_39453569581083
// MI455X (gfx1250) — matrix operand order repaired and checked by reading, not compiled, not run on hardware
//
#include <hip/hip_runtime.h>

typedef __attribute__((ext_vector_type(16))) __bf16 v16bf;
typedef __attribute__((ext_vector_type(8)))  float  v8f;

#define E_DIM 512
#define HEADS 8
#define HD 64
#define PD 32
#define P_CNT 16
#define NB 15           // blocks per dim
#define L_CNT 225       // NB*NB
#define B_CNT 8
#define ROWS_BLK 1800   // B*L
#define ROWS_ALL 1928   // B*L + B*P
#define FV 4096

// ---------------------------------------------------------------------------
// fp32 -> bf16 conversion
__global__ void cvt_f32_bf16(const float* __restrict__ in, __bf16* __restrict__ out, int n) {
    int i = blockIdx.x * 256 + threadIdx.x;
    if (i < n) out[i] = (__bf16)in[i];
}

// ---------------------------------------------------------------------------
// Pool full feature map: (8,64,256,256) -> bf16 pooled map (8,16,128,128)
__global__ void pool_feat(const float* __restrict__ f, __bf16* __restrict__ pm) {
    int i = blockIdx.x * 256 + threadIdx.x;
    if (i >= 2097152) return;
    int x  = i & 127;
    int y  = (i >> 7) & 127;
    int cg = (i >> 14) & 15;
    int b  = i >> 18;
    const float* base = f + (((size_t)b * 64 + cg * 4) * 256 + 2 * y) * 256 + 2 * x;
    float s = 0.f;
#pragma unroll
    for (int dc = 0; dc < 4; ++dc) {
        const float* p = base + dc * 65536;
        s += p[0] + p[1] + p[256] + p[257];
    }
    pm[i] = (__bf16)(s * 0.0625f);
}

// Pool patches (8,16,64,32,32) -> Ybf rows [1800,1928)
__global__ void pool_patch(const float* __restrict__ p, __bf16* __restrict__ Ybf) {
    int i = blockIdx.x * 256 + threadIdx.x;
    if (i >= 128 * FV) return;
    int fidx = i & 4095;
    int r    = i >> 12;
    int ww = fidx & 15, hh = (fidx >> 4) & 15, cg = fidx >> 8;
    const float* base = p + (((size_t)r * 64 + cg * 4) * 32 + 2 * hh) * 32 + 2 * ww;
    float s = 0.f;
#pragma unroll
    for (int dc = 0; dc < 4; ++dc) {
        const float* q = base + dc * 1024;
        s += q[0] + q[1] + q[32] + q[33];
    }
    Ybf[(size_t)(ROWS_BLK + r) * FV + fidx] = (__bf16)(s * 0.0625f);
}

// Gather overlapping blocks from pooled map -> Ybf rows [0,1800)
__global__ void gather_blocks(const __bf16* __restrict__ pm, __bf16* __restrict__ Ybf) {
    int i = blockIdx.x * 256 + threadIdx.x;
    if (i >= ROWS_BLK * FV) return;
    int fidx = i & 4095;
    int r    = i >> 12;
    int b = r / L_CNT, l = r % L_CNT;
    int bi = l / NB, bj = l % NB;
    int ww = fidx & 15, hh = (fidx >> 4) & 15, cg = fidx >> 8;
    Ybf[i] = pm[(((size_t)b * 16 + cg) * 128 + bi * 8 + hh) * 128 + bj * 8 + ww];
}

// ---------------------------------------------------------------------------
// Tiled bf16 WMMA GEMM: C[m,n] = sum_k A[m,k]*W[n,k] + bias[n]
// A: MxK bf16 row-major, W: NxK bf16 row-major, C: MxN fp32.
// Macro tile 128(M) x 64(N), K step 32, 8 waves, wave w -> rows [16w,16w+16).
// Edge handling: A row indices are CLAMPED to M-1 (no exec-mask branching in
// the hot loop); OOB rows only feed accumulator rows that are never stored.
// N is a multiple of 64 in every call.
#define GT_LDA 40
union FragU { v16bf v; uint4 q[2]; };

#if defined(__has_builtin)
#if __has_builtin(__builtin_amdgcn_global_load_async_to_lds_b128)
#define USE_ASYNC_LDS 1
#endif
#endif
#ifndef USE_ASYNC_LDS
#define USE_ASYNC_LDS 0
#endif

#if USE_ASYNC_LDS
typedef int v4i_n __attribute__((__vector_size__(16)));
typedef __attribute__((address_space(1))) v4i_n* g_v4i;
typedef __attribute__((address_space(3))) v4i_n* l_v4i;
static __device__ __forceinline__ void async_cp16(const __bf16* g, __bf16* l) {
    __builtin_amdgcn_global_load_async_to_lds_b128(
        (g_v4i)(void*)const_cast<__bf16*>(g), (l_v4i)(void*)l, 0, 0);
}
#endif

__global__ __launch_bounds__(256)
void gemm_bf16_wmma(const __bf16* __restrict__ A, const __bf16* __restrict__ W,
                    const float* __restrict__ bias, float* __restrict__ C,
                    int M, int N, int K) {
    const int tid    = threadIdx.x;
    const int wave   = tid >> 5;
    const int lane   = tid & 31;
    const int lr     = lane & 15;
    const int lh     = lane >> 4;
    const int tile_n = blockIdx.x * 64;
    const int tile_m = blockIdx.y * 128;

    // per-thread global source pointers (advance by 32 elems per K step)
    const int arow = tid >> 2;        // 0..63
    const int akc  = tid & 3;         // 0..3 (8-elem chunk within 32)
    int gm0 = tile_m + arow;        if (gm0 >= M) gm0 = M - 1;
    int gm1 = tile_m + arow + 64;   if (gm1 >= M) gm1 = M - 1;
    const int gn = tile_n + arow;     // always < N (N multiple of 64)
    const __bf16* gA0 = A + (size_t)gm0 * K + akc * 8;
    const __bf16* gA1 = A + (size_t)gm1 * K + akc * 8;
    const __bf16* gB  = W + (size_t)gn  * K + akc * 8;
    const int la0 = arow * GT_LDA + akc * 8;
    const int la1 = (arow + 64) * GT_LDA + akc * 8;
    const int lb  = arow * GT_LDA + akc * 8;

    v8f acc[4] = {};
    const int nk = K >> 5;

#if USE_ASYNC_LDS
    __shared__ __align__(16) __bf16 sA[2][128 * GT_LDA];
    __shared__ __align__(16) __bf16 sB[2][64 * GT_LDA];
    // prologue: tile 0 into buffer 0
    async_cp16(gA0, &sA[0][la0]);
    async_cp16(gA1, &sA[0][la1]);
    async_cp16(gB,  &sB[0][lb]);
    for (int i = 0; i < nk; ++i) {
        asm volatile("s_wait_asynccnt 0x0" ::: "memory");
        __syncthreads();           // all waves: tile i landed, prev compute done
        if (i + 1 < nk) {          // prefetch tile i+1 (overlaps compute below)
            int nb = (i + 1) & 1;
            const __bf16* pA0 = gA0 + (size_t)(i + 1) * 32;
            const __bf16* pA1 = gA1 + (size_t)(i + 1) * 32;
            const __bf16* pB  = gB  + (size_t)(i + 1) * 32;
            async_cp16(pA0, &sA[nb][la0]);
            async_cp16(pA1, &sA[nb][la1]);
            async_cp16(pB,  &sB[nb][lb]);
        }
        const __bf16* cA = sA[i & 1];
        const __bf16* cB = sB[i & 1];
        FragU af;
        {   // A frag: lane m -> row m, K{0..7,16..23}; lane m+16 -> K{8..15,24..31}
            const __bf16* p = cA + (wave * 16 + lr) * GT_LDA + lh * 8;
            af.q[0] = *(const uint4*)(p);
            af.q[1] = *(const uint4*)(p + 16);
        }
#pragma unroll
        for (int t = 0; t < 4; ++t) {
            // B frag: lane n -> col n, K{0..7,16..23}; lane n+16 -> K{8..15,24..31}
            FragU bf;
            const __bf16* p = cB + (t * 16 + lr) * GT_LDA + lh * 8;
            bf.q[0] = *(const uint4*)(p);
            bf.q[1] = *(const uint4*)(p + 16);
            acc[t] = __builtin_amdgcn_wmma_f32_16x16x32_bf16(
                false, af.v, false, bf.v, (short)0, acc[t], false, false);
        }
    }
#else
    __shared__ __align__(16) __bf16 sA[128 * GT_LDA];
    __shared__ __align__(16) __bf16 sB[64 * GT_LDA];
    for (int i = 0; i < nk; ++i) {
        uint4 va0 = *(const uint4*)(gA0 + (size_t)i * 32);
        uint4 va1 = *(const uint4*)(gA1 + (size_t)i * 32);
        uint4 vb  = *(const uint4*)(gB  + (size_t)i * 32);
        __syncthreads();
        *(uint4*)(&sA[la0]) = va0;
        *(uint4*)(&sA[la1]) = va1;
        *(uint4*)(&sB[lb])  = vb;
        __syncthreads();
        FragU af;
        {
            const __bf16* p = sA + (wave * 16 + lr) * GT_LDA + lh * 8;
            af.q[0] = *(const uint4*)(p);
            af.q[1] = *(const uint4*)(p + 16);
        }
#pragma unroll
        for (int t = 0; t < 4; ++t) {
            FragU bf;
            const __bf16* p = sB + (t * 16 + lr) * GT_LDA + lh * 8;
            bf.q[0] = *(const uint4*)(p);
            bf.q[1] = *(const uint4*)(p + 16);
            acc[t] = __builtin_amdgcn_wmma_f32_16x16x32_bf16(
                false, af.v, false, bf.v, (short)0, acc[t], false, false);
        }
    }
#endif

    // store: VGPR r -> row (wave*16 + 8*lh + r), col = tile_n + t*16 + lr
    float bv[4];
#pragma unroll
    for (int t = 0; t < 4; ++t) bv[t] = bias ? bias[tile_n + t * 16 + lr] : 0.f;
#pragma unroll
    for (int r = 0; r < 8; ++r) {
        int m = tile_m + wave * 16 + lh * 8 + r;
        if (m < M) {
#pragma unroll
            for (int t = 0; t < 4; ++t) {
                int col = tile_n + t * 16 + lr;
                C[(size_t)m * N + col] = acc[t][r] + bv[t];
            }
        }
    }
}

// ---------------------------------------------------------------------------
// LayerNorm over E=512 (optionally adds residual). One block per row.
__global__ __launch_bounds__(256)
void ln_kernel(const float* __restrict__ in, const float* __restrict__ resid,
               const float* __restrict__ g, const float* __restrict__ bta,
               float* __restrict__ outf, __bf16* __restrict__ outb, int rows) {
    int r = blockIdx.x;
    if (r >= rows) return;
    int t = threadIdx.x;
    __shared__ float red[256];
    size_t base = (size_t)r * E_DIM;
    float x0 = in[base + t];
    float x1 = in[base + 256 + t];
    if (resid) { x0 += resid[base + t]; x1 += resid[base + 256 + t]; }
    red[t] = x0 + x1;
    __syncthreads();
    for (int s = 128; s > 0; s >>= 1) { if (t < s) red[t] += red[t + s]; __syncthreads(); }
    float mu = red[0] * (1.f / 512.f);
    __syncthreads();
    float d0 = x0 - mu, d1 = x1 - mu;
    red[t] = d0 * d0 + d1 * d1;
    __syncthreads();
    for (int s = 128; s > 0; s >>= 1) { if (t < s) red[t] += red[t + s]; __syncthreads(); }
    float rstd = rsqrtf(red[0] * (1.f / 512.f) + 1e-5f);
    float y0 = d0 * rstd * g[t] + bta[t];
    float y1 = d1 * rstd * g[256 + t] + bta[256 + t];
    outf[base + t] = y0;
    outf[base + 256 + t] = y1;
    if (outb) { outb[base + t] = (__bf16)y0; outb[base + 256 + t] = (__bf16)y1; }
}

// ---------------------------------------------------------------------------
// Attention: one block per (batch, head). 16 queries x 225 keys, hd=64.
__global__ __launch_bounds__(256)
void attn_kernel(const float* __restrict__ qkv, float* __restrict__ attnout) {
    int b = blockIdx.x / HEADS;
    int h = blockIdx.x % HEADS;
    const int RB = b * L_CNT;
    const int RP = ROWS_BLK + b * P_CNT;
    __shared__ float sq[16 * 64];
    __shared__ float sc[16 * L_CNT];
    int t = threadIdx.x;

    for (int i = t; i < 16 * 64; i += 256) {
        int p = i >> 6, d = i & 63;
        sq[i] = qkv[(size_t)(RP + p) * 1536 + h * HD + d];
    }
    __syncthreads();
    for (int i = t; i < 16 * L_CNT; i += 256) {
        int l = i >> 4, p = i & 15;
        const float* kr = qkv + (size_t)(RB + l) * 1536 + E_DIM + h * HD;
        float s = 0.f;
#pragma unroll 8
        for (int d = 0; d < 64; ++d) s += sq[p * 64 + d] * kr[d];
        sc[p * L_CNT + l] = s * 0.125f;  // 1/sqrt(64)
    }
    __syncthreads();
    if (t < 16) {
        float mx = -1e30f;
        for (int l = 0; l < L_CNT; ++l) mx = fmaxf(mx, sc[t * L_CNT + l]);
        float sum = 0.f;
        for (int l = 0; l < L_CNT; ++l) {
            float e = __expf(sc[t * L_CNT + l] - mx);
            sc[t * L_CNT + l] = e;
            sum += e;
        }
        float inv = 1.f / sum;
        for (int l = 0; l < L_CNT; ++l) sc[t * L_CNT + l] *= inv;
    }
    __syncthreads();
    for (int i = t; i < 16 * 64; i += 256) {
        int p = i >> 6, d = i & 63;
        float a = 0.f;
        for (int l = 0; l < L_CNT; ++l)
            a += sc[p * L_CNT + l] * qkv[(size_t)(RB + l) * 1536 + 2 * E_DIM + h * HD + d];
        attnout[((size_t)b * 16 + p) * E_DIM + h * HD + d] = a;
    }
}

// ---------------------------------------------------------------------------
// x = gelu_exact(mlp) + res ; outs = x @ Wc.T + bc. One block per row (128).
__global__ __launch_bounds__(256)
void mlp_out_kernel(const float* __restrict__ mlp, const float* __restrict__ res,
                    const float* __restrict__ Wc, const float* __restrict__ bc,
                    float* __restrict__ outs) {
    int r = blockIdx.x;
    int t = threadIdx.x;
    __shared__ float sx[E_DIM];
    __shared__ float red[256];
    for (int i = t; i < E_DIM; i += 256) {
        float v = mlp[(size_t)r * E_DIM + i];
        float gl = 0.5f * v * (1.f + erff(v * 0.70710678f));
        sx[i] = gl + res[(size_t)r * E_DIM + i];
    }
    __syncthreads();
    for (int j = 0; j < 2; ++j) {
        float a = 0.f;
        for (int i = t; i < E_DIM; i += 256) a += sx[i] * Wc[j * E_DIM + i];
        red[t] = a;
        __syncthreads();
        for (int s = 128; s > 0; s >>= 1) { if (t < s) red[t] += red[t + s]; __syncthreads(); }
        if (t == 0) outs[r * 2 + j] = red[0] + bc[j];
        __syncthreads();
    }
}

// ---------------------------------------------------------------------------
// Deterministic segment sums: one block per (batch,label); tree reduction.
__global__ __launch_bounds__(256)
void seg_kernel(const float* __restrict__ probs, const int* __restrict__ masks,
                float* __restrict__ sums, float* __restrict__ cnt) {
    int blk = blockIdx.x;            // 0..135
    int b = blk / 17, lab = blk % 17;
    int t = threadIdx.x;
    const int* mrow = masks + (size_t)b * 65536;
    const float* p0 = probs + (size_t)b * 3 * 65536;
    float s0 = 0.f, s1 = 0.f, s2 = 0.f, c = 0.f;
    for (int i = t; i < 65536; i += 256) {
        if (mrow[i] == lab) {
            s0 += p0[i];
            s1 += p0[65536 + i];
            s2 += p0[131072 + i];
            c += 1.f;
        }
    }
    __shared__ float r0[256], r1[256], r2[256], rc[256];
    r0[t] = s0; r1[t] = s1; r2[t] = s2; rc[t] = c;
    __syncthreads();
    for (int s = 128; s > 0; s >>= 1) {
        if (t < s) { r0[t] += r0[t + s]; r1[t] += r1[t + s]; r2[t] += r2[t + s]; rc[t] += rc[t + s]; }
        __syncthreads();
    }
    if (t == 0) {
        sums[blk * 3 + 0] = r0[0];
        sums[blk * 3 + 1] = r1[0];
        sums[blk * 3 + 2] = r2[0];
        cnt[blk] = rc[0];
    }
}

__global__ void newv_kernel(const float* __restrict__ sums, const float* __restrict__ cnt,
                            const float* __restrict__ outs, float* __restrict__ newv) {
    int i = threadIdx.x;
    if (i >= 136) return;
    int b = i / 17, lab = i % 17;
    float* dst = newv + i * 3;
    if (lab == 0) { dst[0] = 0.f; dst[1] = 0.f; dst[2] = 0.f; return; }
    float c = fmaxf(cnt[i], 1.f);
    float m0 = sums[i * 3 + 0] / c + 1e-6f;
    float m1 = sums[i * 3 + 1] / c + 1e-6f;
    float m2 = sums[i * 3 + 2] / c + 1e-6f;
    int p = lab - 1;
    float o0 = outs[(b * 16 + p) * 2 + 0];
    float o1 = outs[(b * 16 + p) * 2 + 1];
    float n0 = m0 / (0.5f * (m1 + m2)) * (0.5f * (o0 + o1));
    dst[0] = n0; dst[1] = o0; dst[2] = o1;
}

__global__ void scatter_kernel(const float* __restrict__ probs, const int* __restrict__ masks,
                               const float* __restrict__ newv, float* __restrict__ out) {
    int i = blockIdx.x * 256 + threadIdx.x;
    if (i >= B_CNT * 65536) return;
    int b = i >> 16, hw = i & 65535;
    int lab = masks[i];
#pragma unroll
    for (int c = 0; c < 3; ++c) {
        float v = (lab > 0) ? newv[(b * 17 + lab) * 3 + c]
                            : probs[((size_t)b * 3 + c) * 65536 + hw];
        out[((size_t)b * 3 + c) * 65536 + hw] = v;
    }
}

// ---------------------------------------------------------------------------
extern "C" void kernel_launch(void* const* d_in, const int* in_sizes, int n_in,
                              void* d_out, int out_size, void* d_ws, size_t ws_size,
                              hipStream_t stream) {
    (void)in_sizes; (void)n_in; (void)out_size; (void)ws_size;
    const float* features = (const float*)d_in[0];
    const float* probs    = (const float*)d_in[1];
    const float* patches  = (const float*)d_in[2];
    const int*   masks    = (const int*)d_in[3];
    const float* ln_g     = (const float*)d_in[4];
    const float* ln_b     = (const float*)d_in[5];
    const float* Wr       = (const float*)d_in[6];
    const float* br       = (const float*)d_in[7];
    const float* Wqkv     = (const float*)d_in[8];
    const float* bqkv     = (const float*)d_in[9];
    const float* Wo       = (const float*)d_in[10];
    const float* bo       = (const float*)d_in[11];
    const float* Wm       = (const float*)d_in[12];
    const float* bm       = (const float*)d_in[13];
    const float* Wc       = (const float*)d_in[14];
    const float* bc       = (const float*)d_in[15];
    float* out = (float*)d_out;

    char* w = (char*)d_ws;
    size_t off = 0;
    auto alloc = [&](size_t bytes) -> void* {
        void* p = w + off;
        off = (off + bytes + 255) & ~(size_t)255;
        return p;
    };
    __bf16* pm      = (__bf16*)alloc(2097152ull * 2);            // pooled map
    __bf16* Ybf     = (__bf16*)alloc((size_t)ROWS_ALL * FV * 2); // gathered rows
    __bf16* Wrbf    = (__bf16*)alloc(2097152ull * 2);
    __bf16* Wqkvbf  = (__bf16*)alloc(786432ull * 2);
    __bf16* Wobf    = (__bf16*)alloc(262144ull * 2);
    __bf16* Wmbf    = (__bf16*)alloc(262144ull * 2);
    float*  embpre  = (float*)alloc((size_t)ROWS_ALL * E_DIM * 4);
    float*  emb     = (float*)alloc((size_t)ROWS_ALL * E_DIM * 4);
    __bf16* embbf   = (__bf16*)alloc((size_t)ROWS_ALL * E_DIM * 2);
    float*  qkv     = (float*)alloc((size_t)ROWS_ALL * 1536 * 4);
    float*  attnout = (float*)alloc(65536ull * 4);
    __bf16* attnbf  = (__bf16*)alloc(65536ull * 2);
    float*  proj    = (float*)alloc(65536ull * 4);
    float*  res     = (float*)alloc(65536ull * 4);
    __bf16* resbf   = (__bf16*)alloc(65536ull * 2);
    float*  mlpb    = (float*)alloc(65536ull * 4);
    float*  outs    = (float*)alloc(256ull * 4);
    float*  sums    = (float*)alloc(408ull * 4);
    float*  cnt     = (float*)alloc(136ull * 4);
    float*  newv    = (float*)alloc(408ull * 4);

    // weight conversions to bf16
    cvt_f32_bf16<<<8192, 256, 0, stream>>>(Wr, Wrbf, 2097152);
    cvt_f32_bf16<<<3072, 256, 0, stream>>>(Wqkv, Wqkvbf, 786432);
    cvt_f32_bf16<<<1024, 256, 0, stream>>>(Wo, Wobf, 262144);
    cvt_f32_bf16<<<1024, 256, 0, stream>>>(Wm, Wmbf, 262144);

    // pooling + gather
    pool_feat<<<8192, 256, 0, stream>>>(features, pm);
    pool_patch<<<2048, 256, 0, stream>>>(patches, Ybf);
    gather_blocks<<<28800, 256, 0, stream>>>(pm, Ybf);

    // embedding GEMM: 1928 x 4096 x 512
    gemm_bf16_wmma<<<dim3(512 / 64, (ROWS_ALL + 127) / 128), 256, 0, stream>>>(
        Ybf, Wrbf, br, embpre, ROWS_ALL, 512, 4096);
    ln_kernel<<<ROWS_ALL, 256, 0, stream>>>(embpre, nullptr, ln_g, ln_b, emb, embbf, ROWS_ALL);

    // QKV GEMM: 1928 x 512 x 1536
    gemm_bf16_wmma<<<dim3(1536 / 64, (ROWS_ALL + 127) / 128), 256, 0, stream>>>(
        embbf, Wqkvbf, bqkv, qkv, ROWS_ALL, 1536, 512);

    // attention
    attn_kernel<<<B_CNT * HEADS, 256, 0, stream>>>(qkv, attnout);

    // output projection + residual + LN
    cvt_f32_bf16<<<256, 256, 0, stream>>>(attnout, attnbf, 65536);
    gemm_bf16_wmma<<<dim3(512 / 64, 1), 256, 0, stream>>>(
        attnbf, Wobf, bo, proj, 128, 512, 512);
    const float* pe = emb + (size_t)ROWS_BLK * E_DIM;
    ln_kernel<<<128, 256, 0, stream>>>(proj, pe, ln_g, ln_b, res, resbf, 128);

    // MLP GEMM + gelu + classifier
    gemm_bf16_wmma<<<dim3(512 / 64, 1), 256, 0, stream>>>(
        resbf, Wmbf, bm, mlpb, 128, 512, 512);
    mlp_out_kernel<<<128, 256, 0, stream>>>(mlpb, res, Wc, bc, outs);

    // segment statistics + final scatter
    seg_kernel<<<136, 256, 0, stream>>>(probs, masks, sums, cnt);
    newv_kernel<<<1, 256, 0, stream>>>(sums, cnt, outs, newv);
    scatter_kernel<<<2048, 256, 0, stream>>>(probs, masks, newv, out);
}
